// MoELayerNative_3719441678903
// MI455X (gfx1250) — compile-verified, matrix operand order checked by reading, not run on hardware
//
#include <hip/hip_runtime.h>
#include <hip/hip_bf16.h>
#include <cstdint>

// ---------- problem dims (from reference) ----------
constexpr int T = 8192;          // B*S = 4*2048 tokens
constexpr int D = 1024;          // model dim
constexpr int F = 4096;          // ffn dim
constexpr int E = 8;             // experts
constexpr int TM = 32;           // token tile per workgroup (2 WMMA M-tiles)
constexpr int NTHREADS = 256;    // 8 waves (wave32)
constexpr int MAXTILES = T / TM; // 256

// ---------- WMMA vector types (CDNA5 gfx1250) ----------
typedef __bf16 v16bf __attribute__((ext_vector_type(16)));
typedef __bf16 v8bf  __attribute__((ext_vector_type(8)));
typedef float  v8f   __attribute__((ext_vector_type(8)));
typedef float  f32x4 __attribute__((ext_vector_type(4)));

// LDS tiles are stored with 16B K-chunks XOR-swizzled by row (chunk ^= row&15):
// row strides (2048B / 8192B) are multiples of 256B, so without the swizzle all
// 16 rows of a fragment hit the same 4 LDS banks (16-way conflict). The swizzle
// spreads the 16 lanes across all 64 banks for both stores and b128 loads.
__device__ __forceinline__ int swz_half(int row, int col, int stride_halves) {
  return row * stride_halves + ((((col >> 3) ^ (row & 15)) << 3) | (col & 7));
}

// A-fragment load from swizzled LDS. c0 = 16B-chunk index (4*kt + hi); the two
// 8-half chunks of a lane's fragment are c0 and c0+2 (== c0^2 since bit1==0).
__device__ __forceinline__ v16bf load_a_swz(const __bf16* __restrict__ rowbase,
                                            int c0, int s) {
  const int cs = c0 ^ s;
  v8bf lo = *(const v8bf*)(rowbase + (cs << 3));
  v8bf hi = *(const v8bf*)(rowbase + ((cs ^ 2) << 3));
  return __builtin_shufflevector(lo, hi, 0,1,2,3,4,5,6,7,8,9,10,11,12,13,14,15);
}

// B-fragment load from global (linear layout), kb = 32*kt + 8*hi halves.
__device__ __forceinline__ v16bf load_b(const __bf16* __restrict__ row, int kb) {
  v8bf lo = *(const v8bf*)(row + kb);
  v8bf hi = *(const v8bf*)(row + kb + 16);
  return __builtin_shufflevector(lo, hi, 0,1,2,3,4,5,6,7,8,9,10,11,12,13,14,15);
}
__device__ __forceinline__ v16bf load_b(const float* __restrict__ row, int kb) {
  f32x4 b0 = *(const f32x4*)(row + kb);
  f32x4 b1 = *(const f32x4*)(row + kb + 4);
  f32x4 b2 = *(const f32x4*)(row + kb + 16);
  f32x4 b3 = *(const f32x4*)(row + kb + 20);
  v16bf b;
#pragma unroll
  for (int q = 0; q < 4; ++q) {
    b[q]      = (__bf16)b0[q];
    b[4 + q]  = (__bf16)b1[q];
    b[8 + q]  = (__bf16)b2[q];
    b[12 + q] = (__bf16)b3[q];
  }
  return b;
}

__device__ __forceinline__ float fast_silu(float v) {
  return v * __builtin_amdgcn_rcpf(1.f + __expf(-v));   // exp + rcp, no IEEE div
}

// K-loop over a 2(M) x 4(N) accumulator block. The 4 B fragments (weights) are
// shared across both M-tiles, halving weight traffic per WMMA vs a 1x4 block.
// Next K-step's fragments are prefetched before the current step's 8 WMMAs.
template<int KT, typename WT>
__device__ __forceinline__ void gemm_tile(const __bf16* __restrict__ arow0,
                                          const __bf16* __restrict__ arow1,
                                          int s, const WT* __restrict__ brow0,
                                          size_t bstride, int hi,
                                          v8f (&cacc)[2][4]) {
  v16bf a_c[2] = { load_a_swz(arow0, hi, s), load_a_swz(arow1, hi, s) };
  v16bf b_c[4];
#pragma unroll
  for (int j = 0; j < 4; ++j) b_c[j] = load_b(brow0 + (size_t)(j * 16) * bstride, hi * 8);

#pragma unroll 2
  for (int kt = 0; kt < KT - 1; ++kt) {
    const int c0n = (kt + 1) * 4 + hi;
    const int kbn = (kt + 1) * 32 + hi * 8;
    v16bf a_n[2] = { load_a_swz(arow0, c0n, s), load_a_swz(arow1, c0n, s) };
    v16bf b_n[4];
#pragma unroll
    for (int j = 0; j < 4; ++j) b_n[j] = load_b(brow0 + (size_t)(j * 16) * bstride, kbn);
#pragma unroll
    for (int j = 0; j < 4; ++j)
#pragma unroll
      for (int mt = 0; mt < 2; ++mt)
        cacc[mt][j] = __builtin_amdgcn_wmma_f32_16x16x32_bf16(
            false, a_c[mt], false, b_c[j], (short)0, cacc[mt][j], false, false);
    a_c[0] = a_n[0]; a_c[1] = a_n[1];
#pragma unroll
    for (int j = 0; j < 4; ++j) b_c[j] = b_n[j];
  }
#pragma unroll
  for (int j = 0; j < 4; ++j)
#pragma unroll
    for (int mt = 0; mt < 2; ++mt)
      cacc[mt][j] = __builtin_amdgcn_wmma_f32_16x16x32_bf16(
          false, a_c[mt], false, b_c[j], (short)0, cacc[mt][j], false, false);
}

// =====================================================================
// Kernel 0: one-time fp32 -> bf16 weight conversion (HBM-rate prepass).
// =====================================================================
__global__ __launch_bounds__(NTHREADS)
void convert_f32_bf16(const float* __restrict__ src, __bf16* __restrict__ dst) {
  const size_t i = ((size_t)blockIdx.x * NTHREADS + threadIdx.x) * 8;
  f32x4 a = *(const f32x4*)(src + i);
  f32x4 b = *(const f32x4*)(src + i + 4);
  v8bf o;
#pragma unroll
  for (int q = 0; q < 4; ++q) { o[q] = (__bf16)a[q]; o[4 + q] = (__bf16)b[q]; }
  *(v8bf*)(dst + i) = o;
}

// =====================================================================
// Kernel 1: gating + top-2 routing. One wave per token.
// =====================================================================
__global__ __launch_bounds__(NTHREADS)
void moe_gate_route(const float* __restrict__ x, const float* __restrict__ gw,
                    int* __restrict__ cnt, int* __restrict__ list,
                    float* __restrict__ wts) {
  __shared__ float sgw[E * D];   // 32 KB
  const int tid = threadIdx.x;
  for (int i = tid; i < E * D; i += NTHREADS) sgw[i] = gw[i];
  __syncthreads();

  const int lane = tid & 31;
  const int wave = tid >> 5;
  const int t = blockIdx.x * 8 + wave;
  const float* xr = x + (size_t)t * D;

  float acc[E];
#pragma unroll
  for (int e = 0; e < E; ++e) acc[e] = 0.f;
  for (int d = lane; d < D; d += 32) {
    const float xv = xr[d];
#pragma unroll
    for (int e = 0; e < E; ++e) acc[e] += xv * sgw[e * D + d];
  }
#pragma unroll
  for (int e = 0; e < E; ++e)
    for (int off = 16; off > 0; off >>= 1)
      acc[e] += __shfl_xor(acc[e], off, 32);

  if (lane == 0) {
    int i1 = 0;
#pragma unroll
    for (int e = 1; e < E; ++e) if (acc[e] > acc[i1]) i1 = e;
    int i2 = (i1 == 0) ? 1 : 0;
#pragma unroll
    for (int e = 0; e < E; ++e) if (e != i1 && acc[e] > acc[i2]) i2 = e;
    const float r2 = __expf(acc[i2] - acc[i1]);   // renormalized top-2 softmax
    const float wA = __builtin_amdgcn_rcpf(1.f + r2);
    const float wB = 1.f - wA;
    int p1 = atomicAdd(&cnt[i1], 1);
    list[i1 * T + p1] = t; wts[i1 * T + p1] = wA;
    int p2 = atomicAdd(&cnt[i2], 1);
    list[i2 * T + p2] = t; wts[i2 * T + p2] = wB;
  }
}

// =====================================================================
// Kernel 2: fused per-expert FFN on gathered 32-token tiles.
//   h = silu(x_tile @ w1[e]^T)  (LDS, bf16, swizzled)
//   out[t,:] += w * (h @ w2[e]^T)
// LDS = 64 KB (x) + 256 KB (h) = 320 KB (CDNA5 per-WG max) -> 1 WG/WGP,
// but each B fragment now feeds 2 M-tiles: weight traffic per FLOP halved.
// =====================================================================
template<typename WT>
__global__ __launch_bounds__(NTHREADS)
void moe_expert_ffn(const float* __restrict__ x, const WT* __restrict__ w1,
                    const WT* __restrict__ w2, const int* __restrict__ cnt,
                    const int* __restrict__ list, const float* __restrict__ wts,
                    float* __restrict__ out) {
  extern __shared__ char smem[];
  __bf16* ldsx = (__bf16*)smem;                  // 32*1024*2 =  64 KB (swizzled)
  __bf16* ldsh = (__bf16*)(smem + TM * D * 2);   // 32*4096*2 = 256 KB (swizzled)

  const int e    = blockIdx.x / MAXTILES;
  const int tile = blockIdx.x % MAXTILES;
  const int nt   = cnt[e];
  if (tile * TM >= nt) return;     // uniform early-exit for empty tiles

  const int tid  = threadIdx.x;
  const int lane = tid & 31;
  const int wave = tid >> 5;
  const int r    = lane & 15;      // fragment row (M for A, N for B/C)
  const int hi   = lane >> 4;      // K-half select
  const int eT   = e * T;

  // ---- stage x tile (fp32 -> bf16, swizzled) ----
  for (int i = tid; i < TM * D; i += NTHREADS) {
    const int row = i >> 10, col = i & (D - 1);
    const int idx = tile * TM + row;
    float v = 0.f;
    if (idx < nt) { const int tk = list[eT + idx]; v = x[(size_t)tk * D + col]; }
    ldsx[swz_half(row, col, D)] = (__bf16)v;
  }
  __syncthreads();

  // ---------------- Phase A: h = silu(x @ w1^T) ----------------
  {
    const WT* w1e = w1 + (size_t)e * F * D;
    const int fbase = wave * (F / 8);          // 512-wide F slice per wave
    for (int nfo = 0; nfo < 32; nfo += 4) {    // 2x4 tile block per step
      const int f0 = fbase + nfo * 16;
      v8f cacc[2][4] = {};
      gemm_tile<D / 32>(ldsx + r * D, ldsx + (r + 16) * D, r & 15,
                        w1e + (size_t)(f0 + r) * D, D, hi, cacc);
      // silu + swizzled store (C layout: m = mt*16 + vgpr + 8*hi, n = r)
#pragma unroll
      for (int mt = 0; mt < 2; ++mt)
#pragma unroll
        for (int j = 0; j < 4; ++j) {
          const int fj = f0 + j * 16;
#pragma unroll
          for (int i2 = 0; i2 < 8; ++i2) {
            const int m = mt * 16 + i2 + hi * 8;
            ldsh[swz_half(m, fj + r, F)] = (__bf16)fast_silu(cacc[mt][j][i2]);
          }
        }
    }
  }
  __syncthreads();

  // ---------------- Phase B: out += w * (h @ w2^T) ----------------
  {
    const WT* w2e = w2 + (size_t)e * D * F;
    const int dbase = wave * (D / 8);          // 128-wide D slice per wave
    // routing info for this lane's 16 C rows (broadcast loads, L2-resident)
    int   tk[2][8];
    float wv[2][8];
#pragma unroll
    for (int mt = 0; mt < 2; ++mt)
#pragma unroll
      for (int i2 = 0; i2 < 8; ++i2) {
        const int idx = tile * TM + mt * 16 + i2 + hi * 8;
        if (idx < nt) { tk[mt][i2] = list[eT + idx]; wv[mt][i2] = wts[eT + idx]; }
        else          { tk[mt][i2] = -1;             wv[mt][i2] = 0.f; }
      }
    for (int ndo = 0; ndo < 8; ndo += 4) {
      const int d0 = dbase + ndo * 16;
      v8f cacc[2][4] = {};
      gemm_tile<F / 32>(ldsh + r * F, ldsh + (r + 16) * F, r & 15,
                        w2e + (size_t)(d0 + r) * F, F, hi, cacc);
      // scale by routing weight, commutative fp32 scatter-add (2 experts/token)
#pragma unroll
      for (int mt = 0; mt < 2; ++mt)
#pragma unroll
        for (int j = 0; j < 4; ++j) {
          const int dj = d0 + j * 16;
#pragma unroll
          for (int i2 = 0; i2 < 8; ++i2) {
            if (tk[mt][i2] >= 0)
              atomicAdd(out + (size_t)tk[mt][i2] * D + dj + r,
                        cacc[mt][j][i2] * wv[mt][i2]);
          }
        }
    }
  }
}

// =====================================================================
// Host entry
// =====================================================================
extern "C" void kernel_launch(void* const* d_in, const int* in_sizes, int n_in,
                              void* d_out, int out_size, void* d_ws, size_t ws_size,
                              hipStream_t stream) {
  const float* x  = (const float*)d_in[0];   // [4,2048,1024]
  const float* gw = (const float*)d_in[1];   // [8,1024]
  const float* w1 = (const float*)d_in[2];   // [8,4096,1024]
  const float* w2 = (const float*)d_in[3];   // [8,1024,4096]
  float* out = (float*)d_out;

  char* ws    = (char*)d_ws;
  int*   cnt  = (int*)ws;                               // E counters
  int*   list = (int*)(ws + 256);                       // E*T token ids
  float* wts  = (float*)(ws + 256 + (size_t)E * T * 4); // E*T weights

  const size_t route_bytes = 1u << 20;                  // 1 MB, aligned region
  const size_t welems      = (size_t)E * F * D;         // 33,554,432 per tensor
  const size_t need_bf16   = route_bytes + 2 * welems * sizeof(__bf16); // ~135 MB

  hipMemsetAsync(ws, 0, 256, stream);                               // zero counters
  hipMemsetAsync(d_out, 0, (size_t)out_size * sizeof(float), stream);

  moe_gate_route<<<dim3(T / 8), dim3(NTHREADS), 0, stream>>>(x, gw, cnt, list, wts);

  const size_t smem_bytes = (size_t)TM * D * 2 + (size_t)TM * F * 2; // 320 KB exact
  const dim3 grid(E * MAXTILES), block(NTHREADS);

  if (ws_size >= need_bf16) {
    // preferred path: bf16 weights staged once in workspace (fits 192 MB L2)
    __bf16* w1b = (__bf16*)(ws + route_bytes);
    __bf16* w2b = w1b + welems;
    const int cblocks = (int)(welems / (8 * NTHREADS));  // exact: 16384
    convert_f32_bf16<<<dim3(cblocks), block, 0, stream>>>(w1, w1b);
    convert_f32_bf16<<<dim3(cblocks), block, 0, stream>>>(w2, w2b);

    hipFuncSetAttribute(reinterpret_cast<const void*>(&moe_expert_ffn<__bf16>),
                        hipFuncAttributeMaxDynamicSharedMemorySize, (int)smem_bytes);
    moe_expert_ffn<__bf16><<<grid, block, smem_bytes, stream>>>(
        x, w1b, w2b, cnt, list, wts, out);
  } else {
    // fallback: stream fp32 weights, convert inline
    hipFuncSetAttribute(reinterpret_cast<const void*>(&moe_expert_ffn<float>),
                        hipFuncAttributeMaxDynamicSharedMemorySize, (int)smem_bytes);
    moe_expert_ffn<float><<<grid, block, smem_bytes, stream>>>(
        x, w1, w2, cnt, list, wts, out);
  }
}
